// AutoEncoder_42356967473707
// MI455X (gfx1250) — matrix operand order repaired and checked by reading, not compiled, not run on hardware
//
#include <hip/hip_runtime.h>
#include <hip/hip_bf16.h>

typedef __attribute__((ext_vector_type(16))) _Float16 v16h;
typedef __attribute__((ext_vector_type(8)))  float    v8f;

// ---------------------------------------------------------------------------
// Encoder conv1: (32,1,256,256) -> (32,16,128,128), k=4 s=2 p=1, raw output
// Cin=1 (16 MACs/output) -> bandwidth-bound, stays scalar.
// ---------------------------------------------------------------------------
__global__ void k_conv1(const float* __restrict__ x, const float* __restrict__ w,
                        const float* __restrict__ bias, float* __restrict__ out) {
  int o = blockIdx.x * 256 + threadIdx.x;            // < 32*16*128*128
  int ox = o & 127, t = o >> 7;
  int oy = t & 127; t >>= 7;
  int co = t & 15;  int n = t >> 4;
  float v = bias[co];
  const float* xp = x + n * 65536;
  __builtin_prefetch(xp + (oy * 2) * 256 + ox * 2, 0, 0);
  #pragma unroll
  for (int ky = 0; ky < 4; ++ky) {
    int iy = oy * 2 - 1 + ky;
    if ((unsigned)iy < 256u) {
      #pragma unroll
      for (int kx = 0; kx < 4; ++kx) {
        int ix = ox * 2 - 1 + kx;
        if ((unsigned)ix < 256u)
          v += xp[iy * 256 + ix] * w[co * 16 + ky * 4 + kx];
      }
    }
  }
  out[o] = v;
}

// ---------------------------------------------------------------------------
// Per-channel batch stats: mean and rsqrt(var+eps) over (N,H,W)
// ---------------------------------------------------------------------------
__global__ void k_stats(const float* __restrict__ buf, int C, int N, int HW,
                        float* __restrict__ mean, float* __restrict__ rstd) {
  int c = blockIdx.x;
  __shared__ float ssum[256], ssq[256];
  int tid = threadIdx.x;
  int total = N * HW;
  float s = 0.f, q = 0.f;
  for (int i = tid; i < total; i += 256) {
    int n = i / HW, p = i - n * HW;
    float v = buf[(n * C + c) * HW + p];
    s += v; q += v * v;
  }
  ssum[tid] = s; ssq[tid] = q;
  __syncthreads();
  for (int off = 128; off; off >>= 1) {
    if (tid < off) { ssum[tid] += ssum[tid + off]; ssq[tid] += ssq[tid + off]; }
    __syncthreads();
  }
  if (tid == 0) {
    float m = ssum[0] / (float)total;
    float var = ssq[0] / (float)total - m * m;
    mean[c] = m;
    rstd[c] = rsqrtf(var + 1e-5f);
  }
}

// ---------------------------------------------------------------------------
// Encoder conv2 as WMMA implicit GEMM.
// (32,16,128,128) bn1+relu on read -> (32,32,64,64) raw.
// M = 131072 pixels, N = 32 (2 tiles), K = 16ci*16taps = 256 = 8 K-steps.
// K order: k = tap*16 + ci, tap = ky*4+kx. 16 WMMAs per 16-row tile.
// ---------------------------------------------------------------------------
__global__ void k_conv2_wmma(const float* __restrict__ in, const float* __restrict__ w,
                             const float* __restrict__ bias,
                             const float* __restrict__ mean, const float* __restrict__ rstd,
                             const float* __restrict__ g, const float* __restrict__ be,
                             float* __restrict__ out) {
  __shared__ float sc[16], sh[16];
  __shared__ _Float16 sB[8][32][32];   // [K-step][K32][co] = 16 KB
  int tid  = threadIdx.x;
  int wave = tid >> 5, lane = tid & 31;
  int grp  = lane >> 4, col = lane & 15;

  if (tid < 16) {
    float r = rstd[tid], m = mean[tid], gg = g[tid];
    sc[tid] = gg * r;
    sh[tid] = be[tid] - m * gg * r;
  }
  for (int idx = tid; idx < 8192; idx += 256) {
    int k = idx >> 5, co = idx & 31;          // k in [0,256)
    int tap = k >> 4, ci = k & 15;
    int ky = tap >> 2, kx = tap & 3;
    sB[k >> 5][k & 31][co] = (_Float16)w[((co * 16 + ci) * 4 + ky) * 4 + kx];
  }
  __syncthreads();

  int tilem = blockIdx.x * 8 + wave;          // [0, 8192)
  int m = tilem * 16 + col;
  int ox = m & 63, oy = (m >> 6) & 63, n = m >> 12;
  const float* ib = in + (n << 18);           // n*16*128*128

  v8f c0, c1;
  float bb0 = bias[col], bb1 = bias[col + 16];
  #pragma unroll
  for (int r = 0; r < 8; ++r) { c0[r] = bb0; c1[r] = bb1; }

  for (int s = 0; s < 8; ++s) {
    v16h a;
    #pragma unroll
    for (int j = 0; j < 16; ++j) {
      int K32 = (j < 8) ? (j + 8 * grp) : (j + 8 + 8 * grp);
      int tap = 2 * s + (K32 >> 4);
      int ci  = K32 & 15;
      int iy = 2 * oy - 1 + (tap >> 2);
      int ix = 2 * ox - 1 + (tap & 3);
      float hv = 0.f;
      if ((unsigned)iy < 128u && (unsigned)ix < 128u)
        hv = fmaxf(ib[(ci << 14) + iy * 128 + ix] * sc[ci] + sh[ci], 0.f);
      a[j] = (_Float16)hv;
    }
    v16h b0, b1;
    #pragma unroll
    for (int j = 0; j < 16; ++j) {
      int K32 = (j < 8) ? (j + 8 * grp) : (j + 8 + 8 * grp);
      b0[j] = sB[s][K32][col];
      b1[j] = sB[s][K32][col + 16];
    }
    c0 = __builtin_amdgcn_wmma_f32_16x16x32_f16(false, a, false, b0, (short)0, c0, false, false);
    c1 = __builtin_amdgcn_wmma_f32_16x16x32_f16(false, a, false, b1, (short)0, c1, false, false);
  }

  #pragma unroll
  for (int r = 0; r < 8; ++r) {
    int mm = tilem * 16 + r + 8 * grp;
    int xx = mm & 63, yy = (mm >> 6) & 63, nn = mm >> 12;
    out[((nn * 32 + col) << 12) + yy * 64 + xx]      = c0[r];
    out[((nn * 32 + col + 16) << 12) + yy * 64 + xx] = c1[r];
  }
}

// ---------------------------------------------------------------------------
// 1x1 conv, optional bn+relu on input (act=1) : Cin=32 -> Cout=32, HW=4096
// 32 MACs/output -> bandwidth-bound, stays scalar.
// ---------------------------------------------------------------------------
template <int ACT>
__global__ void k_conv1x1(const float* __restrict__ in, const float* __restrict__ w,
                          const float* __restrict__ bias,
                          const float* __restrict__ mean, const float* __restrict__ rstd,
                          const float* __restrict__ g, const float* __restrict__ be,
                          float* __restrict__ out) {
  __shared__ float sc[32], sh[32];
  int tid = threadIdx.x;
  if (ACT && tid < 32) {
    float r = rstd[tid], m = mean[tid], gg = g[tid];
    sc[tid] = gg * r;
    sh[tid] = be[tid] - m * gg * r;
  }
  if (ACT) __syncthreads();
  int o = blockIdx.x * 256 + tid;                    // < 32*32*4096
  int s = o & 4095;
  int co = (o >> 12) & 31;
  int n = o >> 17;
  float v = bias[co];
  const float* ip = in + (n << 17) + s;              // n*32*4096
  #pragma unroll 8
  for (int ci = 0; ci < 32; ++ci) {
    float hv = ip[ci << 12];
    if (ACT) hv = fmaxf(hv * sc[ci] + sh[ci], 0.f);
    v += hv * w[co * 32 + ci];
  }
  out[o] = v;
}

// ---------------------------------------------------------------------------
// In-place BN (no relu) for z_e_x, C=32, HW=4096
// ---------------------------------------------------------------------------
__global__ void k_bn_inplace(float* __restrict__ buf,
                             const float* __restrict__ mean, const float* __restrict__ rstd,
                             const float* __restrict__ g, const float* __restrict__ be) {
  __shared__ float sc[32], sh[32];
  int tid = threadIdx.x;
  if (tid < 32) {
    float r = rstd[tid], m = mean[tid], gg = g[tid];
    sc[tid] = gg * r;
    sh[tid] = be[tid] - m * gg * r;
  }
  __syncthreads();
  int o = blockIdx.x * 256 + tid;
  int c = (o >> 12) & 31;
  buf[o] = buf[o] * sc[c] + sh[c];
}

// ---------------------------------------------------------------------------
// VQ: WMMA GEMM zt(131072x32) x emb^T(32x512) with fused argmin + gather.
// Codebook (64 KB f32) staged into LDS with global_load_async_to_lds_b128
// (ASYNCcnt path); -0.5|e|^2 computed on the fly via cross-half shfl.
// Block = 128 threads (4 waves), all lanes active (EXEC all-ones for WMMA).
// ---------------------------------------------------------------------------
__global__ void k_vq(const float* __restrict__ ze, const float* __restrict__ emb,
                     float* __restrict__ zq, float* __restrict__ lat) {
  __shared__ float sembf[512 * 32];     // 64 KB f32 codebook
  int tid  = threadIdx.x;
  int lane = tid & 31, wave = tid >> 5;
  int grp  = lane >> 4, col = lane & 15;

  // ---- async-stage the codebook: 65536 B = 4 waves x 32 iters x 512 B ----
  unsigned ldsbase = (unsigned)(unsigned long long)(uintptr_t)(void*)sembf;
  #pragma unroll 4
  for (int t = 0; t < 32; ++t) {
    unsigned off  = ((unsigned)(wave * 32 + t) << 9) + ((unsigned)lane << 4);
    unsigned lofs = ldsbase + off;
    asm volatile("global_load_async_to_lds_b128 %0, %1, %2"
                 :: "v"(lofs), "v"(off), "s"(emb) : "memory");
  }
  asm volatile("s_wait_asynccnt 0" ::: "memory");
  __syncthreads();

  int tile_m = blockIdx.x * 4 + wave;

  // ---- load A: 16 rows of zt, f32 NCHW gather -> f16, ISA 16x32 layout ----
  int m   = tile_m * 16 + col;
  int n   = m >> 12;                 // / 4096 (H*W)
  int rem = m & 4095;
  const float* zp = ze + (n << 17) + rem;
  v16h a;
  #pragma unroll
  for (int j = 0; j < 16; ++j) {
    int c = (j < 8) ? (j + 8 * grp) : (j + 8 + 8 * grp);
    a[j] = (_Float16)zp[c << 12];
  }

  float bestv[8];
  int   besti[8];
  #pragma unroll
  for (int r = 0; r < 8; ++r) { bestv[r] = -3.0e38f; besti[r] = 0; }

  // ---- sweep 32 N-tiles of 16 codes each ----
  for (int t = 0; t < 32; ++t) {
    int code = t * 16 + col;
    v16h b;
    float sq = 0.f;
    #pragma unroll
    for (int j = 0; j < 16; ++j) {
      float e = sembf[code * 32 + ((j < 8) ? (j + 8 * grp) : (j + 8 + 8 * grp))];
      sq += e * e;
      b[j] = (_Float16)e;
    }
    sq += __shfl_xor(sq, 16, 32);             // combine both channel halves
    float off = -0.5f * sq;
    v8f c;
    #pragma unroll
    for (int r = 0; r < 8; ++r) c[r] = off;
    c = __builtin_amdgcn_wmma_f32_16x16x32_f16(
        false, a, false, b, (short)0, c, false, false);
    #pragma unroll
    for (int r = 0; r < 8; ++r) {
      if (c[r] > bestv[r]) { bestv[r] = c[r]; besti[r] = code; }  // strict > = first-min
    }
  }

  // ---- reduce across the 16 columns of each half-wave ----
  #pragma unroll
  for (int r = 0; r < 8; ++r) {
    for (int o = 8; o >= 1; o >>= 1) {
      float ov = __shfl_xor(bestv[r], o, 32);
      int   oi = __shfl_xor(besti[r], o, 32);
      if (ov > bestv[r] || (ov == bestv[r] && oi < besti[r])) {
        bestv[r] = ov; besti[r] = oi;
      }
    }
  }

  // ---- write latents + gather z_q (f32 codebook from LDS) ----
  #pragma unroll
  for (int r = 0; r < 8; ++r) {
    int row = r + 8 * grp;
    int mm  = tile_m * 16 + row;
    int nn  = mm >> 12, rr = mm & 4095;
    int k   = besti[r];
    float* qp = zq + (nn << 17) + rr;
    qp[col << 12]        = sembf[k * 32 + col];
    qp[(col + 16) << 12] = sembf[k * 32 + col + 16];
    if (col == 0) lat[mm] = (float)k;
  }
}

// ---------------------------------------------------------------------------
// Decoder convT1 as WMMA implicit GEMM.
// ConvT(32->16,k4,s2,p1): (32,32,64,64) bn+relu on read -> (32,16,128,128) raw.
// 4 phases; K = 4taps*32ci = 128 = 4 K-steps, each step = one tap (32 ci).
// N = 16 = one N-tile. 4 WMMAs per 16-row tile.
// Weight layout (Cin=32, Cout=16, 4, 4); ky = (1-py)+2*kyi, iy = y2+py-kyi.
// ---------------------------------------------------------------------------
__global__ void k_dconvT1_wmma(const float* __restrict__ in, const float* __restrict__ w,
                               const float* __restrict__ bias,
                               const float* __restrict__ mean, const float* __restrict__ rstd,
                               const float* __restrict__ g, const float* __restrict__ be,
                               float* __restrict__ out) {
  __shared__ float sc[32], sh[32];
  __shared__ _Float16 sB[4][32][16];   // [tap][ci][co] = 4 KB
  int tid  = threadIdx.x;
  int wave = tid >> 5, lane = tid & 31;
  int grp  = lane >> 4, col = lane & 15;

  unsigned W0 = (unsigned)blockIdx.x * 8u;
  int phase = (int)(W0 >> 13);               // 8192 tiles per phase (8192 % 8 == 0)
  int py = phase >> 1, px = phase & 1;

  if (tid < 32) {
    float r = rstd[tid], m = mean[tid], gg = g[tid];
    sc[tid] = gg * r;
    sh[tid] = be[tid] - m * gg * r;
  }
  for (int idx = tid; idx < 2048; idx += 256) {
    int tap = idx >> 9;
    int ci  = (idx >> 4) & 31;
    int co  = idx & 15;
    int kyi = tap >> 1, kxi = tap & 1;
    int ky = (1 - py) + 2 * kyi;
    int kx = (1 - px) + 2 * kxi;
    sB[tap][ci][co] = (_Float16)w[((ci * 16 + co) * 4 + ky) * 4 + kx];
  }
  __syncthreads();

  unsigned Wid = W0 + (unsigned)wave;
  int tilem = (int)(Wid & 8191u);
  int m  = tilem * 16 + col;
  int x2 = m & 63, y2 = (m >> 6) & 63, n = m >> 12;
  const float* ib = in + (n << 17);          // n*32*64*64

  v8f c;
  float bb = bias[col];
  #pragma unroll
  for (int r = 0; r < 8; ++r) c[r] = bb;

  #pragma unroll
  for (int tap = 0; tap < 4; ++tap) {
    int iy = y2 + py - (tap >> 1);
    int ix = x2 + px - (tap & 1);
    bool inb = ((unsigned)iy < 64u) && ((unsigned)ix < 64u);
    const float* ip = ib + iy * 64 + ix;
    v16h a;
    #pragma unroll
    for (int j = 0; j < 16; ++j) {
      int ci = (j < 8) ? (j + 8 * grp) : (j + 8 + 8 * grp);
      float hv = 0.f;
      if (inb) hv = fmaxf(ip[ci << 12] * sc[ci] + sh[ci], 0.f);
      a[j] = (_Float16)hv;
    }
    v16h b;
    #pragma unroll
    for (int j = 0; j < 16; ++j) b[j] = sB[tap][(j < 8) ? (j + 8 * grp) : (j + 8 + 8 * grp)][col];
    c = __builtin_amdgcn_wmma_f32_16x16x32_f16(false, a, false, b, (short)0, c, false, false);
  }

  #pragma unroll
  for (int r = 0; r < 8; ++r) {
    int mm = tilem * 16 + r + 8 * grp;
    int xx = mm & 63, yy = (mm >> 6) & 63, nn = mm >> 12;
    int oy = 2 * yy + py, ox = 2 * xx + px;
    out[((nn * 16 + col) << 14) + (oy << 7) + ox] = c[r];
  }
}

// ---------------------------------------------------------------------------
// Decoder convT2 as WMMA implicit GEMM.
// ConvT(16->30,k4,s2,p1): (32,16,128,128) -> (32,30,256,256), bn+relu on read.
// Phase (oy&1, ox&1) selects 2x2 taps; K = 4 taps * 16 ci = 64 = 2 WMMA steps;
// N = 30 channels padded to 32 = 2 N-tiles. 4 WMMAs per wave per 16-row tile.
// ---------------------------------------------------------------------------
__global__ void k_dconvT2_wmma(const float* __restrict__ in, const float* __restrict__ w,
                               const float* __restrict__ bias,
                               const float* __restrict__ mean, const float* __restrict__ rstd,
                               const float* __restrict__ g, const float* __restrict__ be,
                               float* __restrict__ out) {
  __shared__ float sc[16], sh[16];
  __shared__ float sbias[32];
  __shared__ _Float16 sB[2][32][32];   // [K-step][K32][co], 4 KB

  int tid  = threadIdx.x;
  int wave = tid >> 5, lane = tid & 31;
  int grp  = lane >> 4, col = lane & 15;

  unsigned W0 = (unsigned)blockIdx.x * 8u;   // first wave id of block
  int phase = (int)(W0 >> 15);               // constant per block (32768 | 8)
  int py = phase >> 1, px = phase & 1;

  if (tid < 16) {
    float r = rstd[tid], m = mean[tid], gg = g[tid];
    sc[tid] = gg * r;
    sh[tid] = be[tid] - m * gg * r;
  }
  if (tid < 32) sbias[tid] = (tid < 30) ? bias[tid] : 0.f;
  for (int idx = tid; idx < 2048; idx += 256) {
    int k = idx >> 5, co = idx & 31;
    int tap = k >> 4, ci = k & 15;
    int kyi = tap >> 1, kxi = tap & 1;
    int ky = (1 - py) + 2 * kyi;
    int kx = (1 - px) + 2 * kxi;
    float wv = (co < 30) ? w[((ci * 30 + co) * 4 + ky) * 4 + kx] : 0.f;
    sB[k >> 5][k & 31][co] = (_Float16)wv;
  }
  __syncthreads();

  unsigned Wid = W0 + (unsigned)wave;
  int tilem = (int)(Wid & 32767u);

  // ---- im2col gather of A for both K-steps (bounds produce zeros) ----
  int m  = tilem * 16 + col;
  int x2 = m & 127, y2 = (m >> 7) & 127, n = m >> 14;
  const float* ib = in + (n << 18);          // n * 16 * 128 * 128
  v16h a0, a1;
  #pragma unroll
  for (int j = 0; j < 16; ++j) {
    int K32 = (j < 8) ? (j + 8 * grp) : (j + 8 + 8 * grp);
    #pragma unroll
    for (int s = 0; s < 2; ++s) {
      int k = 32 * s + K32;
      int tap = k >> 4, ci = k & 15;
      int iy = y2 + py - (tap >> 1);
      int ix = x2 + px - (tap & 1);
      float hv = 0.f;
      if ((unsigned)iy < 128u && (unsigned)ix < 128u)
        hv = fmaxf(ib[(ci << 14) + iy * 128 + ix] * sc[ci] + sh[ci], 0.f);
      if (s == 0) a0[j] = (_Float16)hv; else a1[j] = (_Float16)hv;
    }
  }

  // ---- B vectors from LDS (ISA 32x16 layout) ----
  v16h b00, b01, b10, b11;
  #pragma unroll
  for (int j = 0; j < 16; ++j) {
    int K32 = (j < 8) ? (j + 8 * grp) : (j + 8 + 8 * grp);
    b00[j] = sB[0][K32][col];
    b01[j] = sB[0][K32][col + 16];
    b10[j] = sB[1][K32][col];
    b11[j] = sB[1][K32][col + 16];
  }

  v8f c0, c1;
  float bb0 = sbias[col], bb1 = sbias[col + 16];
  #pragma unroll
  for (int r = 0; r < 8; ++r) { c0[r] = bb0; c1[r] = bb1; }
  c0 = __builtin_amdgcn_wmma_f32_16x16x32_f16(false, a0, false, b00, (short)0, c0, false, false);
  c1 = __builtin_amdgcn_wmma_f32_16x16x32_f16(false, a0, false, b01, (short)0, c1, false, false);
  c0 = __builtin_amdgcn_wmma_f32_16x16x32_f16(false, a1, false, b10, (short)0, c0, false, false);
  c1 = __builtin_amdgcn_wmma_f32_16x16x32_f16(false, a1, false, b11, (short)0, c1, false, false);

  // ---- scatter D to NCHW x_tilde ----
  #pragma unroll
  for (int r = 0; r < 8; ++r) {
    int mm = tilem * 16 + r + 8 * grp;
    int xx = mm & 127, yy = (mm >> 7) & 127, nn = mm >> 14;
    int oy = 2 * yy + py, ox = 2 * xx + px;
    out[((nn * 30 + col) << 16) + (oy << 8) + ox] = c0[r];
    if (col < 14)
      out[((nn * 30 + col + 16) << 16) + (oy << 8) + ox] = c1[r];
  }
}

// ---------------------------------------------------------------------------
extern "C" void kernel_launch(void* const* d_in, const int* in_sizes, int n_in,
                              void* d_out, int out_size, void* d_ws, size_t ws_size,
                              hipStream_t stream) {
  const float* x    = (const float*)d_in[0];
  const float* w1   = (const float*)d_in[1];
  const float* b1   = (const float*)d_in[2];
  const float* g1   = (const float*)d_in[3];
  const float* be1  = (const float*)d_in[4];
  const float* w2   = (const float*)d_in[5];
  const float* b2   = (const float*)d_in[6];
  const float* g2   = (const float*)d_in[7];
  const float* be2  = (const float*)d_in[8];
  const float* w3   = (const float*)d_in[9];
  const float* b3   = (const float*)d_in[10];
  const float* g3   = (const float*)d_in[11];
  const float* be3  = (const float*)d_in[12];
  const float* emb  = (const float*)d_in[13];
  const float* dw1  = (const float*)d_in[14];
  const float* db1  = (const float*)d_in[15];
  const float* dg1  = (const float*)d_in[16];
  const float* dbe1 = (const float*)d_in[17];
  const float* dtw1 = (const float*)d_in[18];
  const float* dtb1 = (const float*)d_in[19];
  const float* dg2  = (const float*)d_in[20];
  const float* dbe2 = (const float*)d_in[21];
  const float* dtw2 = (const float*)d_in[22];
  const float* dtb2 = (const float*)d_in[23];

  float* out = (float*)d_out;
  float* xt  = out;                           // (32,30,256,256)
  float* ze  = out + 62914560;                // (32,32,64,64)
  float* zq  = out + 67108864;                // (32,32,64,64)
  float* lat = out + 71303168;                // (32,64,64)

  float* wsf   = (float*)d_ws;
  float* bufA  = wsf;                         // 8,388,608 floats (h1raw, then dt1raw)
  float* bufB  = wsf + 8388608;               // 4,194,304 floats (h2raw, then d1raw)
  float* stats = wsf + 8388608 + 4194304;     // 256 floats of BN stats
  float* m1 = stats +   0, *r1 = stats +  16; // bn1 (16 ch)
  float* m2 = stats +  32, *r2 = stats +  64; // bn2 (32 ch)
  float* m3 = stats +  96, *r3 = stats + 128; // bn3 (32 ch)
  float* m4 = stats + 160, *r4 = stats + 192; // dec bn1 (32 ch)
  float* m5 = stats + 224, *r5 = stats + 240; // dec bn2 (16 ch)

  // encoder
  k_conv1<<<32768, 256, 0, stream>>>(x, w1, b1, bufA);
  k_stats<<<16, 256, 0, stream>>>(bufA, 16, 32, 16384, m1, r1);
  // conv2 via WMMA implicit GEMM (8192 m-tiles / 8 waves)
  k_conv2_wmma<<<1024, 256, 0, stream>>>(bufA, w2, b2, m1, r1, g1, be1, bufB);
  k_stats<<<32, 256, 0, stream>>>(bufB, 32, 32, 4096, m2, r2);
  k_conv1x1<1><<<16384, 256, 0, stream>>>(bufB, w3, b3, m2, r2, g2, be2, ze);
  k_stats<<<32, 256, 0, stream>>>(ze, 32, 32, 4096, m3, r3);
  k_bn_inplace<<<16384, 256, 0, stream>>>(ze, m3, r3, g3, be3);
  // vector quantization (async-LDS staged codebook + WMMA GEMM + argmin)
  k_vq<<<2048, 128, 0, stream>>>(ze, emb, zq, lat);
  // decoder
  k_conv1x1<0><<<16384, 256, 0, stream>>>(zq, dw1, db1, nullptr, nullptr, nullptr, nullptr, bufB);
  k_stats<<<32, 256, 0, stream>>>(bufB, 32, 32, 4096, m4, r4);
  // convT1 via WMMA implicit GEMM (4 phases x 8192 m-tiles / 8 waves)
  k_dconvT1_wmma<<<4096, 256, 0, stream>>>(bufB, dtw1, dtb1, m4, r4, dg1, dbe1, bufA);
  k_stats<<<16, 256, 0, stream>>>(bufA, 16, 32, 16384, m5, r5);
  // x_tilde via WMMA implicit GEMM (4 phases x 32768 m-tiles / 8 waves)
  k_dconvT2_wmma<<<16384, 256, 0, stream>>>(bufA, dtw2, dtb2, m5, r5, dg2, dbe2, xt);
}
